// EfficientAttention_15461882266051
// MI455X (gfx1250) — compile-verified, matrix operand order checked by reading, not run on hardware
//
#include <hip/hip_runtime.h>

// ---------------------------------------------------------------------------
// EfficientAttention (linear attention) for MI455X / gfx1250.
// bf16 WMMA (v_wmma_f32_16x16x32_bf16) for all matrix math, f32 accumulate,
// fp32 in/out. Vectorized global loads, packed LDS staging, prefetch.
// ---------------------------------------------------------------------------

typedef __attribute__((ext_vector_type(16))) __bf16 v16bf;
typedef __attribute__((ext_vector_type(8)))  __bf16 v8bf;
typedef __attribute__((ext_vector_type(8)))  float  v8f;

#define B_      16
#define C_      128
#define N_      16384      // H*W
#define HEADS_  8
#define DK_     16

// ---- fragment element placement for 16x32 bf16 A/B tiles (ISA 7.12.2) ----
// lanes 0-15: e0..7 -> K 0..7,  e8..15 -> K 16..23
// lanes16-31: e0..7 -> K 8..15, e8..15 -> K 24..31 ; row/col = lane & 15
__device__ __forceinline__ int frag_idx(int row, int k) {
  int lane = (row & 15) + (((k >> 3) & 1) << 4);
  int e    = (k & 7) + ((k & 16) ? 8 : 0);
  return lane * 16 + e;
}

__device__ __forceinline__ v16bf lds_frag(const __bf16* tile, int lane) {
  v16bf a;
  __builtin_memcpy(&a, tile + lane * 16, 32);  // 2x ds_load_b128
  return a;
}

// packed store of 4 bf16 (8 bytes, 8B-aligned in our layouts) -> ds_store_b64
union Pack4 { __bf16 h[4]; uint2 u2; };
__device__ __forceinline__ void store4bf(__bf16* p, float a, float b,
                                         float c, float d) {
  Pack4 t;
  t.h[0] = (__bf16)a; t.h[1] = (__bf16)b; t.h[2] = (__bf16)c; t.h[3] = (__bf16)d;
  *(uint2*)p = t.u2;
}
__device__ __forceinline__ void store4bf_raw(__bf16* p, __bf16 a, __bf16 b,
                                             __bf16 c, __bf16 d) {
  Pack4 t;
  t.h[0] = a; t.h[1] = b; t.h[2] = c; t.h[3] = d;
  *(uint2*)p = t.u2;
}

// ---- float <-> order-preserving uint (for atomic max on floats) ----
__device__ __forceinline__ unsigned ford(float f) {
  unsigned u = __float_as_uint(f);
  return (u & 0x80000000u) ? ~u : (u | 0x80000000u);
}
__device__ __forceinline__ float fdec(unsigned u) {
  return __uint_as_float((u & 0x80000000u) ? (u ^ 0x80000000u) : ~u);
}

// ===========================================================================
// K0: init atomic accumulators (workspace is poisoned by harness)
// ===========================================================================
__global__ void k_init(unsigned* __restrict__ rowmax,
                       float* __restrict__ sumexp,
                       float* __restrict__ ctx) {
  int i = blockIdx.x * blockDim.x + threadIdx.x;
  if (i < B_ * C_) { rowmax[i] = 0u; sumexp[i] = 0.0f; }
  if (i < B_ * HEADS_ * DK_ * DK_) ctx[i] = 0.0f;
}

// ===========================================================================
// K1: fused K/Q/V projection GEMM (bf16 WMMA, K=128) + q channel-softmax +
//     key row-max (atomic). Block = 256 threads (8 waves), covers 512 cols.
// ===========================================================================
__global__ __launch_bounds__(256) void k_qkv(
    const float* __restrict__ x,
    const float* __restrict__ Wk, const float* __restrict__ bk,
    const float* __restrict__ Wq, const float* __restrict__ bq,
    const float* __restrict__ Wv, const float* __restrict__ bv,
    __bf16* __restrict__ keys, __bf16* __restrict__ qbuf,
    __bf16* __restrict__ vbuf, unsigned* __restrict__ rowmax) {
  extern __shared__ char smem[];
  __bf16* wA = (__bf16*)smem;                        // [3][8 mt][4 kt][512]
  __bf16* xB = (__bf16*)(smem + 3 * 8 * 4 * 512 * 2);// [4 kt][8 nt][512]

  const int tid  = threadIdx.x;
  const int wave = tid >> 5, lane = tid & 31;
  const int b    = blockIdx.x >> 5;          // 32 blocks per batch
  const int n0   = (blockIdx.x & 31) * 512;
  const int hi   = (lane >> 4) * 8;          // C-fragment row-half offset
  const int col  = lane & 15;

  // stage three weight matrices once per block: float4 loads, b64 LDS stores
  {
    const float* Ws[3] = {Wk, Wq, Wv};
    for (int m3 = 0; m3 < 3; ++m3) {
      const float* W = Ws[m3];
      for (int i = tid; i < C_ * C_ / 4; i += 256) {
        int m = i >> 5, k4 = (i & 31) * 4;
        float4 f = *(const float4*)(W + m * C_ + k4);
        // 4 consecutive k -> same lane, consecutive e  => contiguous in LDS
        store4bf(wA + ((m3 * 8 + (m >> 4)) * 4 + (k4 >> 5)) * 512 +
                     frag_idx(m & 15, k4 & 31),
                 f.x, f.y, f.z, f.w);
      }
    }
  }

  float bias[3][8];
  for (int r = 0; r < 8; ++r) {
    int row = wave * 16 + hi + r;
    bias[0][r] = bk[row]; bias[1][r] = bq[row]; bias[2][r] = bv[row];
  }

  float kmax[8];
  for (int r = 0; r < 8; ++r) kmax[r] = -3.4e38f;

  for (int sub = 0; sub < 4; ++sub) {
    __syncthreads();
    // stage 128x128 fp32 x-tile: 4x4 register transpose, packed LDS stores
    const float* xp = x + (size_t)b * C_ * N_ + n0 + sub * 128;
    for (int i = tid; i < C_ * 128 / 16; i += 256) {
      int c4 = (i >> 5) * 4, n4 = (i & 31) * 4;
      const float* p = xp + (size_t)c4 * N_ + n4;
      float4 r0 = *(const float4*)(p);
      float4 r1 = *(const float4*)(p + N_);
      float4 r2 = *(const float4*)(p + 2 * N_);
      float4 r3 = *(const float4*)(p + 3 * N_);
      float m[4][4] = {{r0.x, r0.y, r0.z, r0.w}, {r1.x, r1.y, r1.z, r1.w},
                       {r2.x, r2.y, r2.z, r2.w}, {r3.x, r3.y, r3.z, r3.w}};
      int base = ((c4 >> 5) * 8 + (n4 >> 4)) * 512;
      for (int dn = 0; dn < 4; ++dn) {
        int n = n4 + dn;
        // 4 consecutive c -> same lane, consecutive e => contiguous in LDS
        store4bf(xB + base + frag_idx(n & 15, c4 & 31),
                 m[0][dn], m[1][dn], m[2][dn], m[3][dn]);
      }
    }
    // prefetch next sub-tile while this one computes
    if (sub < 3) {
      const float* nxt = x + (size_t)b * C_ * N_ + n0 + (sub + 1) * 128;
      __builtin_prefetch(nxt + (size_t)(tid >> 1) * N_ + (tid & 1) * 64, 0, 1);
    }
    __syncthreads();

    for (int m3 = 0; m3 < 3; ++m3) {
      v8f acc[8];
      for (int nt = 0; nt < 8; ++nt)
        for (int r = 0; r < 8; ++r) acc[nt][r] = 0.0f;

      for (int kt = 0; kt < 4; ++kt) {
        v16bf a = lds_frag(wA + ((m3 * 8 + wave) * 4 + kt) * 512, lane);
        for (int nt = 0; nt < 8; ++nt) {
          v16bf bb = lds_frag(xB + (kt * 8 + nt) * 512, lane);
          acc[nt] = __builtin_amdgcn_wmma_f32_16x16x32_bf16(
              false, a, false, bb, (short)0, acc[nt], false, false);
        }
      }

      if (m3 == 0) {          // keys: store bf16, track row max
        for (int nt = 0; nt < 8; ++nt) {
          int n = n0 + sub * 128 + nt * 16 + col;
          for (int r = 0; r < 8; ++r) {
            float f = acc[nt][r] + bias[0][r];
            int row = wave * 16 + hi + r;
            keys[((size_t)(b * C_ + row)) * N_ + n] = (__bf16)f;
            kmax[r] = fmaxf(kmax[r], f);
          }
        }
      } else if (m3 == 1) {   // queries: softmax over 16 head-channels
        for (int nt = 0; nt < 8; ++nt) {
          int n = n0 + sub * 128 + nt * 16 + col;
          float vals[8], m8 = -3.4e38f;
          for (int r = 0; r < 8; ++r) {
            vals[r] = acc[nt][r] + bias[1][r];
            m8 = fmaxf(m8, vals[r]);
          }
          float m = fmaxf(m8, __shfl_xor(m8, 16, 32));  // pair lane-halves
          float s8 = 0.0f;
          for (int r = 0; r < 8; ++r) { vals[r] = __expf(vals[r] - m); s8 += vals[r]; }
          float inv = 1.0f / (s8 + __shfl_xor(s8, 16, 32));
          for (int r = 0; r < 8; ++r) {
            int row = wave * 16 + hi + r;
            qbuf[((size_t)(b * C_ + row)) * N_ + n] = (__bf16)(vals[r] * inv);
          }
        }
      } else {                // values: store bf16
        for (int nt = 0; nt < 8; ++nt) {
          int n = n0 + sub * 128 + nt * 16 + col;
          for (int r = 0; r < 8; ++r) {
            int row = wave * 16 + hi + r;
            vbuf[((size_t)(b * C_ + row)) * N_ + n] = (__bf16)(acc[nt][r] + bias[2][r]);
          }
        }
      }
    }
  }

  // reduce key row-max across the 16 lanes of each row-half, then atomic
  for (int off = 1; off < 16; off <<= 1)
    for (int r = 0; r < 8; ++r)
      kmax[r] = fmaxf(kmax[r], __shfl_xor(kmax[r], off, 32));
  if (col == 0)
    for (int r = 0; r < 8; ++r)
      atomicMax(rowmax + b * C_ + wave * 16 + hi + r, ford(kmax[r]));
}

// ===========================================================================
// K2: context[b,h] += exp(keys - rowmax) @ v^T  (WMMA K=32 slices) and
//     sumexp row reduction. grid = (B*H, N/2048), block = 256.
//     Per-lane fragment K-runs are contiguous -> 16B vector loads.
// ===========================================================================
__global__ __launch_bounds__(256) void k_ctx(
    const __bf16* __restrict__ keys, const __bf16* __restrict__ vbuf,
    const unsigned* __restrict__ rowmax,
    float* __restrict__ ctx, float* __restrict__ sumexp) {
  __shared__ float red[8 * 256];
  __shared__ float sred[8 * 16];

  const int tid = threadIdx.x, wave = tid >> 5, lane = tid & 31;
  const int bh = blockIdx.x;                 // 0..127
  const int b = bh >> 3, h = bh & 7;
  const int nbase = blockIdx.y * 2048;
  const int myrow = lane & 15;
  const int khalf = (lane >> 4) ? 8 : 0;

  const __bf16* krow = keys + ((size_t)(b * C_ + h * DK_) + myrow) * N_;
  const __bf16* vrow = vbuf + ((size_t)(b * C_ + h * DK_) + myrow) * N_;
  const float rmax = fdec(rowmax[b * C_ + h * DK_ + myrow]);

  v8f acc; for (int r = 0; r < 8; ++r) acc[r] = 0.0f;
  float sacc = 0.0f;

  for (int it = wave; it < 64; it += 8) {    // 2048 cols / 32 per slice
    int n0 = nbase + it * 32;
    const __bf16* kb = krow + n0 + khalf;
    const __bf16* vb = vrow + n0 + khalf;
    v8bf k0 = *(const v8bf*)(kb);
    v8bf k1 = *(const v8bf*)(kb + 16);
    v8bf w0 = *(const v8bf*)(vb);
    v8bf w1 = *(const v8bf*)(vb + 16);
    __builtin_prefetch(kb + 256, 0, 0);      // next slice for this wave
    __builtin_prefetch(vb + 256, 0, 0);

    v16bf a;
    for (int e = 0; e < 8; ++e) {
      float f0 = __expf((float)k0[e] - rmax); sacc += f0; a[e]     = (__bf16)f0;
      float f1 = __expf((float)k1[e] - rmax); sacc += f1; a[e + 8] = (__bf16)f1;
    }
    v16bf bb = __builtin_shufflevector(w0, w1, 0, 1, 2, 3, 4, 5, 6, 7,
                                       8, 9, 10, 11, 12, 13, 14, 15);
    acc = __builtin_amdgcn_wmma_f32_16x16x32_bf16(
        false, a, false, bb, (short)0, acc, false, false);
  }

  for (int r = 0; r < 8; ++r) red[wave * 256 + lane * 8 + r] = acc[r];
  float s2 = sacc + __shfl_xor(sacc, 16, 32);
  if (lane < 16) sred[wave * 16 + lane] = s2;
  __syncthreads();

  {  // one atomicAdd per thread into the (kc,vc) context tile
    float t = 0.0f;
    for (int w = 0; w < 8; ++w) t += red[w * 256 + tid];
    int l = tid >> 3, r = tid & 7;
    int kc = r + ((l >= 16) ? 8 : 0), vc = l & 15;
    atomicAdd(ctx + bh * 256 + kc * 16 + vc, t);
  }
  if (tid < 16) {
    float t = 0.0f;
    for (int w = 0; w < 8; ++w) t += sred[w * 16 + tid];
    atomicAdd(sumexp + bh * 16 + tid, t);
  }
}

// ===========================================================================
// K3: Wfused[b][c][h*16+kc] = sum_v Wr[c, h*16+v] * ctx[b,h,kc,v]/sumexp
// ===========================================================================
__global__ void k_fuse(const float* __restrict__ Wr,
                       const float* __restrict__ ctx,
                       const float* __restrict__ sumexp,
                       float* __restrict__ Wfused) {
  int i = blockIdx.x * blockDim.x + threadIdx.x;
  if (i >= B_ * C_ * C_) return;
  int b = i >> 14, c = (i >> 7) & 127, kcol = i & 127;
  int h = kcol >> 4, kc = kcol & 15;
  const float* cx = ctx + ((b * 8 + h) * 16 + kc) * 16;
  const float* wr = Wr + c * C_ + h * 16;
  float inv = 1.0f / sumexp[(b * 8 + h) * 16 + kc];
  float acc = 0.0f;
  for (int v = 0; v < 16; ++v) acc += wr[v] * cx[v];
  Wfused[i] = acc * inv;
}

// ===========================================================================
// K4: out = Wfused[b] @ q + br + x   (bf16 WMMA, K=128, fp32 epilogue)
// ===========================================================================
__global__ __launch_bounds__(256) void k_out(
    const float* __restrict__ Wfused, const __bf16* __restrict__ qbuf,
    const float* __restrict__ br, const float* __restrict__ x,
    float* __restrict__ out) {
  extern __shared__ char smem[];
  __bf16* wA = (__bf16*)smem;                    // [8 mt][4 kt][512]
  __bf16* qB = (__bf16*)(smem + 8 * 4 * 512 * 2);// [4 kt][8 nt][512]

  const int tid  = threadIdx.x;
  const int wave = tid >> 5, lane = tid & 31;
  const int b    = blockIdx.x >> 5;
  const int n0   = (blockIdx.x & 31) * 512;
  const int hi   = (lane >> 4) * 8;
  const int col  = lane & 15;

  const float* W = Wfused + (size_t)b * C_ * C_;
  for (int i = tid; i < C_ * C_ / 4; i += 256) {
    int m = i >> 5, k4 = (i & 31) * 4;
    float4 f = *(const float4*)(W + m * C_ + k4);
    store4bf(wA + ((m >> 4) * 4 + (k4 >> 5)) * 512 + frag_idx(m & 15, k4 & 31),
             f.x, f.y, f.z, f.w);
  }

  float biasr[8];
  for (int r = 0; r < 8; ++r) biasr[r] = br[wave * 16 + hi + r];

  for (int sub = 0; sub < 4; ++sub) {
    __syncthreads();
    // stage q tile (bf16): 4 rows x 8 cols blocks, uint4 loads, b64 stores
    const __bf16* qp = qbuf + (size_t)b * C_ * N_ + n0 + sub * 128;
    for (int i = tid; i < C_ * 128 / 32; i += 256) {
      int c4 = (i >> 4) * 4, n8 = (i & 15) * 8;
      union { uint4 u; __bf16 h[8]; } r0, r1, r2, r3;
      const __bf16* p = qp + (size_t)c4 * N_ + n8;
      r0.u = *(const uint4*)(p);
      r1.u = *(const uint4*)(p + N_);
      r2.u = *(const uint4*)(p + 2 * N_);
      r3.u = *(const uint4*)(p + 3 * N_);
      int base = (c4 >> 5) * 8 * 512;
      for (int dn = 0; dn < 8; ++dn) {
        int n = n8 + dn;
        store4bf_raw(qB + base + (n >> 4) * 512 + frag_idx(n & 15, c4 & 31),
                     r0.h[dn], r1.h[dn], r2.h[dn], r3.h[dn]);
      }
    }
    if (sub < 3) {
      const __bf16* nxt = qbuf + (size_t)b * C_ * N_ + n0 + (sub + 1) * 128;
      __builtin_prefetch(nxt + (size_t)(tid >> 1) * N_ + (tid & 1) * 64, 0, 1);
    }
    __syncthreads();

    v8f acc[8];
    for (int nt = 0; nt < 8; ++nt)
      for (int r = 0; r < 8; ++r) acc[nt][r] = 0.0f;

    for (int kt = 0; kt < 4; ++kt) {
      v16bf a = lds_frag(wA + (wave * 4 + kt) * 512, lane);
      for (int nt = 0; nt < 8; ++nt) {
        v16bf bb = lds_frag(qB + (kt * 8 + nt) * 512, lane);
        acc[nt] = __builtin_amdgcn_wmma_f32_16x16x32_bf16(
            false, a, false, bb, (short)0, acc[nt], false, false);
      }
    }

    for (int nt = 0; nt < 8; ++nt) {
      int n = n0 + sub * 128 + nt * 16 + col;
      for (int r = 0; r < 8; ++r) {
        int row = wave * 16 + hi + r;
        size_t idx = ((size_t)(b * C_ + row)) * N_ + n;
        out[idx] = acc[nt][r] + biasr[r] + x[idx];
      }
    }
  }
}

// ===========================================================================
extern "C" void kernel_launch(void* const* d_in, const int* in_sizes, int n_in,
                              void* d_out, int out_size, void* d_ws, size_t ws_size,
                              hipStream_t stream) {
  (void)in_sizes; (void)n_in; (void)out_size; (void)ws_size;
  const float* x  = (const float*)d_in[0];
  const float* Wk = (const float*)d_in[1];
  const float* bk = (const float*)d_in[2];
  const float* Wq = (const float*)d_in[3];
  const float* bq = (const float*)d_in[4];
  const float* Wv = (const float*)d_in[5];
  const float* bv = (const float*)d_in[6];
  const float* Wr = (const float*)d_in[7];
  const float* br = (const float*)d_in[8];
  float* out = (float*)d_out;

  char* ws = (char*)d_ws;
  const size_t sz_buf = (size_t)B_ * C_ * N_ * 2;   // 64 MB each (bf16)
  __bf16*   keys   = (__bf16*)(ws);
  __bf16*   qbuf   = (__bf16*)(ws + sz_buf);
  __bf16*   vbuf   = (__bf16*)(ws + 2 * sz_buf);
  unsigned* rowmax = (unsigned*)(ws + 3 * sz_buf);
  float*    sumexp = (float*)(ws + 3 * sz_buf + 8192);
  float*    ctx    = (float*)(ws + 3 * sz_buf + 16384);
  float*    Wfused = (float*)(ws + 3 * sz_buf + 16384 + 131072);

  k_init<<<128, 256, 0, stream>>>(rowmax, sumexp, ctx);
  k_qkv<<<512, 256, 131072, stream>>>(x, Wk, bk, Wq, bq, Wv, bv,
                                      keys, qbuf, vbuf, rowmax);
  k_ctx<<<dim3(128, 8), 256, 0, stream>>>(keys, vbuf, rowmax, ctx, sumexp);
  k_fuse<<<1024, 256, 0, stream>>>(Wr, ctx, sumexp, Wfused);
  k_out<<<512, 256, 65536, stream>>>(Wfused, qbuf, br, x, out);
}
